// SingleAttention_39187281608797
// MI455X (gfx1250) — matrix operand order repaired and checked by reading, not compiled, not run on hardware
//
#include <hip/hip_runtime.h>
#include <hip/hip_bf16.h>
#include <math.h>

typedef __attribute__((ext_vector_type(16))) __bf16 v16bf;
typedef __attribute__((ext_vector_type(8)))  __bf16 v8bf;
typedef __attribute__((ext_vector_type(8)))  float  v8f;

#define B_DIM   4
#define SQ_DIM  2048
#define SKV_DIM 2048
#define E_DIM   1024
#define R_DIM   128

static __device__ __forceinline__ __bf16 f2bf(float f) { return (__bf16)f; }

// ---------------------------------------------------------------------------
// One-time weight prep: Wt(R x E) bf16 = transpose(W(E x R)) * scale.
// ---------------------------------------------------------------------------
__global__ __launch_bounds__(256)
void prep_w_kernel(const float* __restrict__ W, __bf16* __restrict__ Wt, float scale)
{
    const int i = blockIdx.x * blockDim.x + threadIdx.x;   // over E*R
    if (i < E_DIM * R_DIM) {
        const int e = i / R_DIM, r = i % R_DIM;
        Wt[(size_t)r * E_DIM + e] = f2bf(W[i] * scale);
    }
}

// ---------------------------------------------------------------------------
// Projection: out(M x 128) = X(M x 1024) @ Wt^T, bf16 WMMA, f32 acc.
// One wave computes a 16x128 strip. All 8 B-fragments of a k-step are loaded
// up front (16 x b128 in one clause) so the WMMA chain runs with staggered
// loadcnt waits instead of a full-latency stall per WMMA.
// ---------------------------------------------------------------------------
__global__ __launch_bounds__(256)
void proj_kernel(const float* __restrict__ X, const __bf16* __restrict__ Wt,
                 __bf16* __restrict__ out, int transpose)
{
    const int lane = threadIdx.x & 31;
    const int wave = (blockIdx.x * blockDim.x + threadIdx.x) >> 5;
    const int m0   = wave * 16;
    const int half = lane >> 4;
    const int l16  = lane & 15;

    v8f acc[8];
#pragma unroll
    for (int j = 0; j < 8; ++j) acc[j] = (v8f){0.f,0.f,0.f,0.f,0.f,0.f,0.f,0.f};

    const int rowA = m0 + l16;
    const __bf16* wbase = Wt + (size_t)l16 * E_DIM + half * 16;

    for (int e0 = 0; e0 < E_DIM; e0 += 32) {
        // Issue all 8 B-fragment loads first (immediate offsets off one base)
        v16bf bfr[8];
#pragma unroll
        for (int j = 0; j < 8; ++j)
            bfr[j] = *(const v16bf*)(wbase + (size_t)j * 16 * E_DIM + e0);

        // A fragment: row rowA, K = e0 + half*16 + {0..15}; converts
        // overlap with the B loads in flight.
        const float4* xp = (const float4*)(X + (size_t)rowA * E_DIM + e0 + half * 16);
        const float4 f0 = xp[0], f1 = xp[1], f2 = xp[2], f3 = xp[3];
        v16bf a;
        a[0]=f2bf(f0.x); a[1]=f2bf(f0.y); a[2]=f2bf(f0.z); a[3]=f2bf(f0.w);
        a[4]=f2bf(f1.x); a[5]=f2bf(f1.y); a[6]=f2bf(f1.z); a[7]=f2bf(f1.w);
        a[8]=f2bf(f2.x); a[9]=f2bf(f2.y); a[10]=f2bf(f2.z); a[11]=f2bf(f2.w);
        a[12]=f2bf(f3.x); a[13]=f2bf(f3.y); a[14]=f2bf(f3.z); a[15]=f2bf(f3.w);

#pragma unroll
        for (int j = 0; j < 8; ++j)
            acc[j] = __builtin_amdgcn_wmma_f32_16x16x32_bf16(
                false, a, false, bfr[j], (short)0, acc[j], false, false);
    }

#pragma unroll
    for (int j = 0; j < 8; ++j) {
        const int n = j * 16 + l16;
#pragma unroll
        for (int r = 0; r < 8; ++r) {
            const int m = m0 + r + 8 * half;
            const float v = acc[j][r];
            if (!transpose) {
                out[(size_t)m * R_DIM + n] = f2bf(v);
            } else {
                const int b_ = m / SKV_DIM;
                const int s  = m % SKV_DIM;
                out[((size_t)b_ * R_DIM + n) * SKV_DIM + s] = f2bf(v);
            }
        }
    }
}

// ---------------------------------------------------------------------------
// Flash attention. Block = 4 waves sharing one batch + 64 consecutive q rows;
// each 32-wide k/v tile is cooperatively staged into LDS once per block
// (4x cut in L2 traffic), fragments then read via ds_load. Online softmax.
// ---------------------------------------------------------------------------
__global__ __launch_bounds__(128)
void attn_kernel(const __bf16* __restrict__ qb, const __bf16* __restrict__ kb,
                 const __bf16* __restrict__ vt, float* __restrict__ out)
{
    __shared__ __bf16 kbuf[32 * 128];      // [kv_local][r]   8 KB
    __shared__ __bf16 vbuf[128 * 32];      // [r][kv_local]   8 KB
    __shared__ __bf16 pbuf[4][16 * 32];    // per-wave P tile 4 KB

    const int t    = threadIdx.x;          // 0..127
    const int lane = t & 31;
    const int wv   = t >> 5;
    const int b    = blockIdx.x / (SQ_DIM / 64);
    const int q0   = (blockIdx.x % (SQ_DIM / 64)) * 64 + wv * 16;
    const int half = lane >> 4;
    const int l16  = lane & 15;

    // Per-thread cooperative-load source pointers (advance by 32 kv per iter)
    const __bf16* ksrc = kb + (size_t)(b * SKV_DIM + (t >> 2)) * R_DIM + (t & 3) * 32;
    const __bf16* vsrc = vt + (size_t)(b * R_DIM + t) * SKV_DIM;
    __bf16* kdst = &kbuf[(t >> 2) * 128 + (t & 3) * 32];
    __bf16* vdst = &vbuf[t * 32];

    // q A-fragments for the 4 R-chunks of 32 (loaded once, reused all iters)
    v16bf qf[4];
    {
        const __bf16* qp = qb + (size_t)(b * SQ_DIM + q0 + l16) * R_DIM + half * 16;
#pragma unroll
        for (int c = 0; c < 4; ++c) {
            v8bf lo = *(const v8bf*)(qp + c * 32);
            v8bf hi = *(const v8bf*)(qp + c * 32 + 8);
#pragma unroll
            for (int i = 0; i < 8; ++i) { qf[c][i] = lo[i]; qf[c][8 + i] = hi[i]; }
        }
    }

    v8f o[8];
#pragma unroll
    for (int j = 0; j < 8; ++j) o[j] = (v8f){0.f,0.f,0.f,0.f,0.f,0.f,0.f,0.f};
    float mrow[8], lrow[8];
#pragma unroll
    for (int r = 0; r < 8; ++r) { mrow[r] = -1e30f; lrow[r] = 0.f; }

    for (int kv0 = 0; kv0 < SKV_DIM; kv0 += 32) {
        // ---- stage k (32x128) and v^T (128x32) tiles into LDS ----
        {
            const __bf16* ks = ksrc + (size_t)kv0 * R_DIM;
            *(v16bf*)(kdst)      = *(const v16bf*)(ks);
            *(v16bf*)(kdst + 16) = *(const v16bf*)(ks + 16);
            const __bf16* vs = vsrc + kv0;
            *(v16bf*)(vdst)      = *(const v16bf*)(vs);
            *(v16bf*)(vdst + 16) = *(const v16bf*)(vs + 16);
        }
        // prefetch next tile while this one is consumed
        if (kv0 + 32 < SKV_DIM) {
            __builtin_prefetch(ksrc + (size_t)(kv0 + 32) * R_DIM, 0, 0);
            __builtin_prefetch(vsrc + kv0 + 32, 0, 0);
        }
        __syncthreads();

        // ---- S = q @ k^T for 32 kv positions (two 16-wide N tiles) ----
        v8f s0 = (v8f){0.f,0.f,0.f,0.f,0.f,0.f,0.f,0.f};
        v8f s1 = (v8f){0.f,0.f,0.f,0.f,0.f,0.f,0.f,0.f};
#pragma unroll
        for (int c = 0; c < 4; ++c) {
            v16bf bk0 = *(const v16bf*)&kbuf[l16 * 128 + c * 32 + half * 16];
            s0 = __builtin_amdgcn_wmma_f32_16x16x32_bf16(
                false, qf[c], false, bk0, (short)0, s0, false, false);
            v16bf bk1 = *(const v16bf*)&kbuf[(16 + l16) * 128 + c * 32 + half * 16];
            s1 = __builtin_amdgcn_wmma_f32_16x16x32_bf16(
                false, qf[c], false, bk1, (short)0, s1, false, false);
        }

        // ---- online softmax update over the 32 new columns ----
        float p0[8], p1[8];
#pragma unroll
        for (int r = 0; r < 8; ++r) {
            float mx = fmaxf(s0[r], s1[r]);
#pragma unroll
            for (int d = 1; d < 16; d <<= 1) mx = fmaxf(mx, __shfl_xor(mx, d, 32));
            const float nm   = fmaxf(mrow[r], mx);
            const float corr = __expf(mrow[r] - nm);
            const float e0v  = __expf(s0[r] - nm);
            const float e1v  = __expf(s1[r] - nm);
            float sum = e0v + e1v;
#pragma unroll
            for (int d = 1; d < 16; d <<= 1) sum += __shfl_xor(sum, d, 32);
            lrow[r] = lrow[r] * corr + sum;
            mrow[r] = nm;
            p0[r] = e0v; p1[r] = e1v;
#pragma unroll
            for (int j = 0; j < 8; ++j) o[j][r] = o[j][r] * corr;
        }

        // ---- C-layout -> A-layout for P via LDS ----
#pragma unroll
        for (int r = 0; r < 8; ++r) {
            const int row = r + 8 * half;
            pbuf[wv][row * 32 + l16]      = f2bf(p0[r]);
            pbuf[wv][row * 32 + 16 + l16] = f2bf(p1[r]);
        }
        __syncthreads();
        v16bf a;
        {
            const __bf16* pp = &pbuf[wv][l16 * 32 + half * 16];
#pragma unroll
            for (int i = 0; i < 8; ++i) { a[i] = pp[i]; a[8 + i] = pp[8 + i]; }
        }

        // ---- O += P @ V (v^T tile in LDS: contiguous B-operand) ----
#pragma unroll
        for (int j = 0; j < 8; ++j) {
            v16bf bv = *(const v16bf*)&vbuf[(j * 16 + l16) * 32 + half * 16];
            o[j] = __builtin_amdgcn_wmma_f32_16x16x32_bf16(
                false, a, false, bv, (short)0, o[j], false, false);
        }
        __syncthreads();   // all waves done with kbuf/vbuf/pbuf before refill
    }

    // ---- epilogue: normalize and store f32 ----
#pragma unroll
    for (int r = 0; r < 8; ++r) {
        const float inv = 1.0f / lrow[r];
        const int m = q0 + r + 8 * half;
#pragma unroll
        for (int j = 0; j < 8; ++j) {
            out[(size_t)(b * SQ_DIM + m) * R_DIM + j * 16 + l16] = o[j][r] * inv;
        }
    }
}

// ---------------------------------------------------------------------------
extern "C" void kernel_launch(void* const* d_in, const int* in_sizes, int n_in,
                              void* d_out, int out_size, void* d_ws, size_t ws_size,
                              hipStream_t stream)
{
    const float* xq  = (const float*)d_in[0];   // (B,SQ,E)
    const float* xkv = (const float*)d_in[1];   // (B,SKV,E)
    const float* Wq  = (const float*)d_in[2];   // (1,E,R)
    const float* Wk  = (const float*)d_in[3];
    const float* Wv  = (const float*)d_in[4];
    float* out = (float*)d_out;                 // (B,SQ,R)

    char* ws = (char*)d_ws;
    const size_t seg  = (size_t)B_DIM * SQ_DIM * R_DIM * 2;  // 2 MB (bf16)
    const size_t wseg = (size_t)E_DIM * R_DIM * 2;           // 256 KB (bf16)
    __bf16* qb  = (__bf16*)(ws);
    __bf16* kbp = (__bf16*)(ws + seg);
    __bf16* vtp = (__bf16*)(ws + 2 * seg);
    __bf16* Wtq = (__bf16*)(ws + 3 * seg);
    __bf16* Wtk = (__bf16*)(ws + 3 * seg + wseg);
    __bf16* Wtv = (__bf16*)(ws + 3 * seg + 2 * wseg);

    const float scale = 0.08838834764831845f;   // 1/sqrt(R), folded into Wq

    prep_w_kernel<<<(E_DIM * R_DIM) / 256, 256, 0, stream>>>(Wq, Wtq, scale);
    prep_w_kernel<<<(E_DIM * R_DIM) / 256, 256, 0, stream>>>(Wk, Wtk, 1.0f);
    prep_w_kernel<<<(E_DIM * R_DIM) / 256, 256, 0, stream>>>(Wv, Wtv, 1.0f);

    // 8192 rows / 16 per wave / 8 waves per block = 64 blocks
    proj_kernel<<<64, 256, 0, stream>>>(xq,  Wtq, qb,  0);
    proj_kernel<<<64, 256, 0, stream>>>(xkv, Wtk, kbp, 0);
    proj_kernel<<<64, 256, 0, stream>>>(xkv, Wtv, vtp, 1);

    // B * SQ/64 = 128 blocks of 4 waves (64 q rows each)
    attn_kernel<<<128, 128, 0, stream>>>(qb, kbp, vtp, out);
}
